// MATGCNBlock_41729902247948
// MI455X (gfx1250) — matrix operand order repaired and checked by reading, not compiled, not run on hardware
//
#include <hip/hip_runtime.h>
#include <math.h>
#include <stdint.h>

#define Bc   16
#define CIN  32
#define COUT 64
#define Nn   1024
#define Tt   24
#define NT   (Nn * Tt)   // 24576

typedef __attribute__((ext_vector_type(16))) _Float16 v16h;
typedef __attribute__((ext_vector_type(8)))  float    v8f;

// ---------------------------------------------------------------------------
// CDNA5 async global->LDS copy (ASYNCcnt path, ISA 10. / 15.18.3 op 98) and
// its completion wait. LDS byte address = low 32 bits of flat shared pointer.
// ---------------------------------------------------------------------------
__device__ __forceinline__ void async_copy_b128(void* lds_dst, const void* gsrc) {
  const uint32_t dst = (uint32_t)(uintptr_t)lds_dst;
  asm volatile("global_load_async_to_lds_b128 %0, %1, off"
               :
               : "v"(dst), "v"(gsrc)
               : "memory");
}
__device__ __forceinline__ void wait_async0() {
  asm volatile("s_wait_asynccnt 0" ::: "memory");
}

// ---------------------------------------------------------------------------
// Generic per-wave 16x16-tile GEMM on v_wmma_f32_16x16x32_f16.
// f32 gathered via lambdas, converted to f16, f32 accumulation.
// Lane layouts per CDNA5 ISA 7.12.2:
//   A (16-bit 16x32): lane = hi*16+lo, row M=lo; VGPR v<4 -> K=2v+{0,1}+hi*8,
//                     v>=4 -> K=16+2(v-4)+{0,1}+hi*8
//   B (16-bit 32x16): col N=lo; element i -> K = i + hi*8 (i<8), 8 + i + hi*8 (i>=8)
//   C/D (f32 16x16):  col N=lo; VGPR r -> M = r + 8*hi
// ---------------------------------------------------------------------------
template <class LA, class LB, class SC>
__device__ __forceinline__ void wmma_gemm_wave(int tm, int tn, int M, int Ncols,
                                               int K, LA loadA, LB loadB, SC store) {
  const int lane = threadIdx.x & 31;
  const int hi   = lane >> 4;
  const int lo   = lane & 15;
  v8f acc = {};
  for (int k0 = 0; k0 < K; k0 += 32) {
    v16h a, b;
    const int mA = tm * 16 + lo;
    const int nB = tn * 16 + lo;
#pragma unroll
    for (int i = 0; i < 16; ++i) {
      const int v  = i >> 1;
      const int kl = (v < 4 ? 2 * v : 16 + 2 * (v - 4)) + (i & 1) + hi * 8;
      const int k  = k0 + kl;
      a[i] = (_Float16)((mA < M && k < K) ? loadA(mA, k) : 0.0f);
    }
#pragma unroll
    for (int i = 0; i < 16; ++i) {
      const int k = k0 + (i < 8 ? i : 8 + i) + hi * 8;
      b[i] = (_Float16)((nB < Ncols && k < K) ? loadB(k, nB) : 0.0f);
    }
    acc = __builtin_amdgcn_wmma_f32_16x16x32_f16(false, a, false, b,
                                                 (short)0, acc, false, false);
  }
  const int n = tn * 16 + lo;
#pragma unroll
  for (int r = 0; r < 8; ++r) {
    const int m = tm * 16 + r + hi * 8;
    if (m < M && n < Ncols) store(m, n, acc[r]);
  }
}

// ---------------------------------------------------------------------------
// 1) LayerNorm over (N,T) per (b,c), affine weights indexed [N,T].
// ---------------------------------------------------------------------------
__global__ void k_layernorm(const float* __restrict__ x, const float* __restrict__ w,
                            const float* __restrict__ bb, float* __restrict__ h0) {
  __shared__ float s_sum[256], s_sq[256];
  const int bc = blockIdx.x;
  const size_t base = (size_t)bc * NT;
  float s = 0.f, sq = 0.f;
  for (int i = threadIdx.x; i < NT; i += 256) {
    const float v = x[base + i];
    s += v; sq += v * v;
  }
  s_sum[threadIdx.x] = s; s_sq[threadIdx.x] = sq;
  __syncthreads();
  for (int off = 128; off > 0; off >>= 1) {
    if (threadIdx.x < off) {
      s_sum[threadIdx.x] += s_sum[threadIdx.x + off];
      s_sq[threadIdx.x]  += s_sq[threadIdx.x + off];
    }
    __syncthreads();
  }
  const float mean = s_sum[0] * (1.0f / NT);
  const float var  = s_sq[0] * (1.0f / NT) - mean * mean;
  const float rstd = rsqrtf(var + 1e-5f);
  for (int i = threadIdx.x; i < NT; i += 256)
    h0[base + i] = (x[base + i] - mean) * rstd * w[i] + bb[i];
}

// ---------------------------------------------------------------------------
// 2) Kc[b,c,t] = sum_n h0[b,c,n,t] * ca_alpha[n]
// ---------------------------------------------------------------------------
__global__ void k_ca_kc(const float* __restrict__ h0, const float* __restrict__ alpha,
                        float* __restrict__ Kc) {
  __shared__ float s[Tt];
  const int bc = blockIdx.x;
  const size_t base = (size_t)bc * NT;
  float p[Tt];
#pragma unroll
  for (int t = 0; t < Tt; ++t) p[t] = 0.f;
  for (int n = threadIdx.x; n < Nn; n += 256) {
    const float al = alpha[n];
    const float* row = h0 + base + (size_t)n * Tt;
#pragma unroll
    for (int t = 0; t < Tt; ++t) p[t] += al * row[t];
  }
  if (threadIdx.x < Tt) s[threadIdx.x] = 0.f;
  __syncthreads();
#pragma unroll
  for (int t = 0; t < Tt; ++t) atomicAdd(&s[t], p[t]);
  __syncthreads();
  if (threadIdx.x < Tt) Kc[bc * Tt + threadIdx.x] = s[threadIdx.x];
}

// ---------------------------------------------------------------------------
// 3) Ac[b,c,d] = softmax_d( sum_{t,s} Kc[b,c,t] caW[t,s] Kc[b,d,s] )
// ---------------------------------------------------------------------------
__global__ void k_ca_softmax(const float* __restrict__ Kc, const float* __restrict__ caW,
                             float* __restrict__ Ac) {
  const int b = blockIdx.x, c = threadIdx.x;
  const float* kcb = Kc + (size_t)b * CIN * Tt;
  float kc[Tt];
#pragma unroll
  for (int t = 0; t < Tt; ++t) kc[t] = kcb[c * Tt + t];
  float m1[Tt];
  for (int s = 0; s < Tt; ++s) {
    float a = 0.f;
    for (int t = 0; t < Tt; ++t) a += kc[t] * caW[t * Tt + s];
    m1[s] = a;
  }
  float lg[CIN], mx = -1e30f;
  for (int d = 0; d < CIN; ++d) {
    float a = 0.f;
    for (int s = 0; s < Tt; ++s) a += m1[s] * kcb[d * Tt + s];
    lg[d] = a; mx = fmaxf(mx, a);
  }
  float sum = 0.f;
  for (int d = 0; d < CIN; ++d) { lg[d] = __expf(lg[d] - mx); sum += lg[d]; }
  const float inv = 1.f / sum;
  for (int d = 0; d < CIN; ++d) Ac[((size_t)b * CIN + c) * CIN + d] = lg[d] * inv;
}

// ---------------------------------------------------------------------------
// 4) h1[b,c,(n,t)] = Ac[b] (32x32) @ h0[b] (32 x 24576)   [WMMA]
// ---------------------------------------------------------------------------
__global__ void k_ca_apply(const float* __restrict__ Ac, const float* __restrict__ h0,
                           float* __restrict__ h1) {
  const int b = blockIdx.y;
  const int wave = (int)((blockIdx.x * blockDim.x + threadIdx.x) >> 5);
  const int tiles_m = CIN / 16;          // 2
  const int tm = wave % tiles_m, tn = wave / tiles_m;
  const float* Ab = Ac + (size_t)b * CIN * CIN;
  const float* Bb = h0 + (size_t)b * CIN * NT;
  float* Cb = h1 + (size_t)b * CIN * NT;
  wmma_gemm_wave(tm, tn, CIN, NT, CIN,
      [=](int m, int k) { return Ab[m * CIN + k]; },
      [=](int k, int n) { return Bb[(size_t)k * NT + n]; },
      [=](int m, int n, float v) { Cb[(size_t)m * NT + n] = v; });
}

// ---------------------------------------------------------------------------
// 5) Ks[b,n,t] = sum_c h1[b,c,n,t] sa_alpha[c]
// ---------------------------------------------------------------------------
__global__ void k_sa_ks(const float* __restrict__ h1, const float* __restrict__ alpha,
                        float* __restrict__ Ks) {
  const int idx = blockIdx.x * 256 + threadIdx.x;
  if (idx >= Bc * NT) return;
  const int r = idx % NT, b = idx / NT;
  const float* p = h1 + (size_t)b * CIN * NT + r;
  float a = 0.f;
  for (int c = 0; c < CIN; ++c) a += alpha[c] * p[(size_t)c * NT];
  Ks[idx] = a;
}

// ---------------------------------------------------------------------------
// 6) KsW[b,n,s] = sum_t Ks[b,n,t] saW[t,s]
// ---------------------------------------------------------------------------
__global__ void k_sa_ksw(const float* __restrict__ Ks, const float* __restrict__ saW,
                         float* __restrict__ KsW) {
  const int idx = blockIdx.x * 256 + threadIdx.x;
  if (idx >= Bc * Nn * Tt) return;
  const int s = idx % Tt;
  const int bn = idx / Tt;
  const float* krow = Ks + (size_t)bn * Tt;
  float a = 0.f;
  for (int t = 0; t < Tt; ++t) a += krow[t] * saW[t * Tt + s];
  KsW[idx] = a;
}

// ---------------------------------------------------------------------------
// 7) logits[b,n,m] = KsW[b] (1024x24) @ Ks[b]^T (24x1024)   [WMMA, K padded]
// ---------------------------------------------------------------------------
__global__ void k_sa_logits(const float* __restrict__ KsW, const float* __restrict__ Ks,
                            float* __restrict__ logits) {
  const int b = blockIdx.y;
  const int wave = (int)((blockIdx.x * blockDim.x + threadIdx.x) >> 5);
  const int tiles_m = Nn / 16;           // 64
  const int tm = wave % tiles_m, tn = wave / tiles_m;
  const float* Ab = KsW + (size_t)b * Nn * Tt;
  const float* Bb = Ks + (size_t)b * Nn * Tt;
  float* Cb = logits + (size_t)b * Nn * Nn;
  wmma_gemm_wave(tm, tn, Nn, Nn, Tt,
      [=](int m, int k) { return Ab[m * Tt + k]; },
      [=](int k, int n) { return Bb[n * Tt + k]; },
      [=](int m, int n, float v) { Cb[(size_t)m * Nn + n] = v; });
}

// ---------------------------------------------------------------------------
// 8) As[b,n,m] = A[n,m] * softmax_m(logits[b,n,m])  (in place on logits)
// ---------------------------------------------------------------------------
__global__ void k_sa_softmax(float* __restrict__ logits, const float* __restrict__ A) {
  __shared__ float red[256];
  const int tid = threadIdx.x;
  const int row = blockIdx.x;            // b*Nn + n
  const int n = row & (Nn - 1);
  float* p = logits + (size_t)row * Nn;
  float mx = -1e30f;
  for (int i = tid; i < Nn; i += 256) mx = fmaxf(mx, p[i]);
  red[tid] = mx; __syncthreads();
  for (int off = 128; off > 0; off >>= 1) {
    if (tid < off) red[tid] = fmaxf(red[tid], red[tid + off]);
    __syncthreads();
  }
  mx = red[0]; __syncthreads();
  float s = 0.f;
  for (int i = tid; i < Nn; i += 256) s += __expf(p[i] - mx);
  red[tid] = s; __syncthreads();
  for (int off = 128; off > 0; off >>= 1) {
    if (tid < off) red[tid] += red[tid + off];
    __syncthreads();
  }
  const float inv = 1.f / red[0];
  const float* Arow = A + (size_t)n * Nn;
  for (int i = tid; i < Nn; i += 256) p[i] = Arow[i] * __expf(p[i] - mx) * inv;
}

// ---------------------------------------------------------------------------
// 9) hT[b,m,c,t] = h1[b,c,m,t]
// ---------------------------------------------------------------------------
__global__ void k_transpose(const float* __restrict__ h1, float* __restrict__ hT) {
  const long long idx = (long long)blockIdx.x * 256 + threadIdx.x;
  if (idx >= (long long)Bc * CIN * NT) return;
  const int t = (int)(idx % Tt);
  const long long r = idx / Tt;
  const int n = (int)(r % Nn);
  const long long bc = r / Nn;
  const int c = (int)(bc % CIN);
  const int b = (int)(bc / CIN);
  hT[(((long long)b * Nn + n) * CIN + c) * Tt + t] = h1[idx];
}

// ---------------------------------------------------------------------------
// 10) g[b,n,(c,t)] = As[b] (1024x1024) @ hT[b] (1024x768)
//     Dominant GEMM. 128x64 block tile, async global->LDS double buffering
//     (ASYNCcnt), 8 waves x (2x2) 16x16 WMMA tiles, 4x LDS reuse per wave.
// ---------------------------------------------------------------------------
#define AGG_BM 128
#define AGG_BN 64
#define AGG_BK 32

__global__ __launch_bounds__(256) void k_gcn_agg(const float* __restrict__ As_,
                                                 const float* __restrict__ hT,
                                                 float* __restrict__ g) {
  __shared__ __align__(16) float sA[2][AGG_BM][AGG_BK];  // 2 x 16 KB
  __shared__ __align__(16) float sB[2][AGG_BK][AGG_BN];  // 2 x  8 KB
  const int b  = blockIdx.z;
  const int m0 = blockIdx.x * AGG_BM;
  const int n0 = blockIdx.y * AGG_BN;
  const float* Ab = As_ + (size_t)b * Nn * Nn;
  const float* Bb = hT + (size_t)b * Nn * (CIN * Tt);
  float* Cb = g + (size_t)b * Nn * (CIN * Tt);

  const int tid  = threadIdx.x;
  const int lane = tid & 31;
  const int wave = tid >> 5;
  const int wm = wave & 3;   // 4 wave-rows  x 32
  const int wn = wave >> 2;  // 2 wave-cols  x 32
  const int hi = lane >> 4, lo = lane & 15;

  // stage one K-slice into LDS buffer `buf` with async b128 copies
  auto stage = [&](int k0, int buf) {
    // A tile: 128x32 floats = 1024 16B-chunks, 4 per thread
#pragma unroll
    for (int j = 0; j < 4; ++j) {
      const int chunk = tid + 256 * j;        // 0..1023
      const int row = chunk >> 3;             // 8 chunks per 32-float row
      const int kc  = (chunk & 7) * 4;
      async_copy_b128(&sA[buf][row][kc], Ab + (size_t)(m0 + row) * Nn + k0 + kc);
    }
    // B tile: 32x64 floats = 512 16B-chunks, 2 per thread
#pragma unroll
    for (int j = 0; j < 2; ++j) {
      const int chunk = tid + 256 * j;        // 0..511
      const int row = chunk >> 4;             // 16 chunks per 64-float row
      const int nc  = (chunk & 15) * 4;
      async_copy_b128(&sB[buf][row][nc], Bb + (size_t)(k0 + row) * (CIN * Tt) + n0 + nc);
    }
  };

  v8f acc[2][2] = {};
  stage(0, 0);
  for (int k0 = 0; k0 < Nn; k0 += AGG_BK) {
    const int buf = (k0 / AGG_BK) & 1;
    wait_async0();
    __syncthreads();                          // staged tile visible; prev compute done
    if (k0 + AGG_BK < Nn) stage(k0 + AGG_BK, buf ^ 1);

    v16h afrag[2], bfrag[2];
#pragma unroll
    for (int mt = 0; mt < 2; ++mt) {
      const int mrow = wm * 32 + mt * 16 + lo;
#pragma unroll
      for (int i = 0; i < 16; ++i) {
        const int v  = i >> 1;
        const int kl = (v < 4 ? 2 * v : 16 + 2 * (v - 4)) + (i & 1) + hi * 8;
        afrag[mt][i] = (_Float16)sA[buf][mrow][kl];
      }
    }
#pragma unroll
    for (int nt = 0; nt < 2; ++nt) {
      const int ncol = wn * 32 + nt * 16 + lo;
#pragma unroll
      for (int i = 0; i < 16; ++i)
        bfrag[nt][i] = (_Float16)sB[buf][(i < 8 ? i : 8 + i) + hi * 8][ncol];
    }
#pragma unroll
    for (int mt = 0; mt < 2; ++mt)
#pragma unroll
      for (int nt = 0; nt < 2; ++nt)
        acc[mt][nt] = __builtin_amdgcn_wmma_f32_16x16x32_f16(
            false, afrag[mt], false, bfrag[nt], (short)0, acc[mt][nt], false, false);
  }

#pragma unroll
  for (int mt = 0; mt < 2; ++mt) {
#pragma unroll
    for (int nt = 0; nt < 2; ++nt) {
      const int n = n0 + wn * 32 + nt * 16 + lo;
#pragma unroll
      for (int r = 0; r < 8; ++r) {
        const int m = m0 + wm * 32 + mt * 16 + r + hi * 8;
        Cb[(size_t)m * (CIN * Tt) + n] = acc[mt][nt][r];
      }
    }
  }
}

// ---------------------------------------------------------------------------
// 11) h2[b,o,n,t] = sum_c gcn_W[o,c] g[b,n,c,t]   [WMMA, batched over (b,t)]
// ---------------------------------------------------------------------------
__global__ void k_gcn_mix(const float* __restrict__ W, const float* __restrict__ g,
                          float* __restrict__ h2) {
  const int b = blockIdx.y, t = blockIdx.z;
  const int wave = (int)((blockIdx.x * blockDim.x + threadIdx.x) >> 5);
  const int tiles_m = COUT / 16;         // 4
  const int tm = wave % tiles_m, tn = wave / tiles_m;
  const float* Bb = g + (size_t)b * Nn * CIN * Tt + t;
  float* Cb = h2 + (size_t)b * COUT * NT + t;
  wmma_gemm_wave(tm, tn, COUT, Nn, CIN,
      [=](int m, int k) { return W[m * CIN + k]; },
      [=](int k, int n) { return Bb[(size_t)n * (CIN * Tt) + (size_t)k * Tt]; },
      [=](int m, int n, float v) { Cb[(size_t)m * NT + (size_t)n * Tt] = v; });
}

// ---------------------------------------------------------------------------
// 12) Kt[b,t,n] = sum_c h2[b,c,n,t] ta_alpha[c]
// ---------------------------------------------------------------------------
__global__ void k_ta_kt(const float* __restrict__ h2, const float* __restrict__ alpha,
                        float* __restrict__ Kt) {
  const int idx = blockIdx.x * 256 + threadIdx.x;
  if (idx >= Bc * Tt * Nn) return;
  const int n = idx % Nn;
  const int bt = idx / Nn;
  const int t = bt % Tt;
  const int b = bt / Tt;
  const float* p = h2 + (size_t)b * COUT * NT + (size_t)n * Tt + t;
  float a = 0.f;
  for (int c = 0; c < COUT; ++c) a += alpha[c] * p[(size_t)c * NT];
  Kt[idx] = a;
}

// ---------------------------------------------------------------------------
// 13) P1[b,t,k] = Kt[b,t,:] . ta_W1[k,:];  P2 likewise
// ---------------------------------------------------------------------------
__global__ void k_ta_p(const float* __restrict__ Kt, const float* __restrict__ W1,
                       const float* __restrict__ W2, float* __restrict__ P1,
                       float* __restrict__ P2) {
  const int idx = blockIdx.x * 256 + threadIdx.x;
  if (idx >= Bc * Tt * 10) return;
  const int k = idx % 10;
  const int bt = idx / 10;
  const float* kt = Kt + (size_t)bt * Nn;
  const float* w1 = W1 + (size_t)k * Nn;
  const float* w2 = W2 + (size_t)k * Nn;
  float a1 = 0.f, a2 = 0.f;
  for (int n = 0; n < Nn; ++n) { a1 += kt[n] * w1[n]; a2 += kt[n] * w2[n]; }
  P1[idx] = a1; P2[idx] = a2;
}

// ---------------------------------------------------------------------------
// 14) At[b,t,s] = softmax_s( P1[b,t,:] . P2[b,s,:] )
// ---------------------------------------------------------------------------
__global__ void k_ta_softmax(const float* __restrict__ P1, const float* __restrict__ P2,
                             float* __restrict__ At) {
  const int b = blockIdx.x, t = threadIdx.x;
  if (t >= Tt) return;
  const float* p1 = P1 + (size_t)(b * Tt + t) * 10;
  float lg[Tt], mx = -1e30f;
  for (int s = 0; s < Tt; ++s) {
    const float* p2 = P2 + (size_t)(b * Tt + s) * 10;
    float a = 0.f;
    for (int k = 0; k < 10; ++k) a += p1[k] * p2[k];
    lg[s] = a; mx = fmaxf(mx, a);
  }
  float sum = 0.f;
  for (int s = 0; s < Tt; ++s) { lg[s] = __expf(lg[s] - mx); sum += lg[s]; }
  const float inv = 1.f / sum;
  for (int s = 0; s < Tt; ++s) At[(size_t)(b * Tt + t) * Tt + s] = lg[s] * inv;
}

// ---------------------------------------------------------------------------
// 15) h3[b,c,n,t] = sum_s At[b,t,s] h2[b,c,n,s]   [WMMA, M=K=24 padded]
// ---------------------------------------------------------------------------
__global__ void k_ta_apply(const float* __restrict__ At, const float* __restrict__ h2,
                           float* __restrict__ h3) {
  const int b = blockIdx.y;
  const int wave = (int)((blockIdx.x * blockDim.x + threadIdx.x) >> 5);
  const int tiles_m = 2;
  const int tm = wave % tiles_m, tn = wave / tiles_m;
  const float* Ab = At + (size_t)b * Tt * Tt;
  const float* Bb = h2 + (size_t)b * COUT * NT;
  float* Cb = h3 + (size_t)b * COUT * NT;
  wmma_gemm_wave(tm, tn, Tt, COUT * Nn, Tt,
      [=](int m, int k) { return Ab[m * Tt + k]; },
      [=](int k, int j) { return Bb[(size_t)(j >> 10) * NT + (size_t)(j & 1023) * Tt + k]; },
      [=](int m, int j, float v) {
        Cb[(size_t)(j >> 10) * NT + (size_t)(j & 1023) * Tt + m] = v;
      });
}

// ---------------------------------------------------------------------------
// 16) causal dilated conv: out[t] = relu(bias + sum_c w1*h[t] + w0*h[t-d])
// ---------------------------------------------------------------------------
__global__ void k_conv(const float* __restrict__ hin, const float* __restrict__ w,
                       const float* __restrict__ bias, float* __restrict__ hout, int dil) {
  const long long idx = (long long)blockIdx.x * 256 + threadIdx.x;
  if (idx >= (long long)Bc * COUT * NT) return;
  const int t = (int)(idx % Tt);
  const long long nt = idx / Tt;
  const int n = (int)(nt % Nn);
  const long long on = nt / Nn;
  const int o = (int)(on % COUT);
  const int b = (int)(on / COUT);
  float acc = bias[o];
  const float* hb = hin + (long long)b * COUT * NT + (long long)n * Tt;
  const float* wo = w + (size_t)o * COUT * 2;
  for (int c = 0; c < COUT; ++c) {
    const float* hc = hb + (long long)c * NT;
    float a = wo[2 * c + 1] * hc[t];
    if (t >= dil) a += wo[2 * c] * hc[t - dil];
    acc += a;
  }
  hout[idx] = fmaxf(acc, 0.f);
}

// ---------------------------------------------------------------------------
// 17) out = relu(h + res_w @ x + res_b)
// ---------------------------------------------------------------------------
__global__ void k_residual(const float* __restrict__ h, const float* __restrict__ x,
                           const float* __restrict__ rw, const float* __restrict__ rb,
                           float* __restrict__ out) {
  const long long idx = (long long)blockIdx.x * 256 + threadIdx.x;
  if (idx >= (long long)Bc * COUT * NT) return;
  const int t = (int)(idx % Tt);
  const long long nt = idx / Tt;
  const int n = (int)(nt % Nn);
  const long long on = nt / Nn;
  const int o = (int)(on % COUT);
  const int b = (int)(on / COUT);
  float acc = h[idx] + rb[o];
  const float* xb = x + (long long)b * CIN * NT + (long long)n * Tt + t;
  const float* w = rw + (size_t)o * CIN;
  for (int c = 0; c < CIN; ++c) acc += w[c] * xb[(long long)c * NT];
  out[idx] = fmaxf(acc, 0.f);
}

// ---------------------------------------------------------------------------
extern "C" void kernel_launch(void* const* d_in, const int* in_sizes, int n_in,
                              void* d_out, int out_size, void* d_ws, size_t ws_size,
                              hipStream_t stream) {
  const float* x        = (const float*)d_in[0];
  const float* A        = (const float*)d_in[1];
  const float* ln_w     = (const float*)d_in[2];
  const float* ln_b     = (const float*)d_in[3];
  const float* ca_W     = (const float*)d_in[4];
  const float* ca_alpha = (const float*)d_in[5];
  const float* sa_W     = (const float*)d_in[6];
  const float* sa_alpha = (const float*)d_in[7];
  const float* gcn_W    = (const float*)d_in[8];
  const float* ta_W1    = (const float*)d_in[9];
  const float* ta_W2    = (const float*)d_in[10];
  const float* ta_alpha = (const float*)d_in[11];
  const float* c1w      = (const float*)d_in[12];
  const float* c1b      = (const float*)d_in[13];
  const float* c2w      = (const float*)d_in[14];
  const float* c2b      = (const float*)d_in[15];
  const float* rw       = (const float*)d_in[16];
  const float* rb       = (const float*)d_in[17];
  float* out = (float*)d_out;
  (void)in_sizes; (void)n_in; (void)out_size; (void)ws_size;

  float* ws = (float*)d_ws;
  size_t cur = 0;
  const size_t SZ_CNT = (size_t)Bc * CIN * NT;    // 12.58M floats
  const size_t SZ_ONT = (size_t)Bc * COUT * NT;   // 25.17M floats
  float* h0  = ws + cur; cur += SZ_CNT;           // reused as hT
  float* h1  = ws + cur; cur += SZ_CNT;           // reused as g
  float* h2  = ws + cur; cur += SZ_ONT;           // reused as conv1 out
  float* h3  = ws + cur; cur += SZ_ONT;           // reused as conv2 out
  float* As  = ws + cur; cur += (size_t)Bc * Nn * Nn;
  float* Kc  = ws + cur; cur += (size_t)Bc * CIN * Tt;
  float* Ac  = ws + cur; cur += (size_t)Bc * CIN * CIN;
  float* Ks  = ws + cur; cur += (size_t)Bc * Nn * Tt;
  float* KsW = ws + cur; cur += (size_t)Bc * Nn * Tt;
  float* Kt  = ws + cur; cur += (size_t)Bc * Tt * Nn;
  float* P1  = ws + cur; cur += (size_t)Bc * Tt * 10;
  float* P2  = ws + cur; cur += (size_t)Bc * Tt * 10;
  float* At  = ws + cur; cur += (size_t)Bc * Tt * Tt;
  float* hT = h0;  // h0 dead after k_ca_apply
  float* g  = h1;  // h1 dead after k_transpose
  float* h4 = h2;  // h2 dead after k_ta_apply
  float* h5 = h3;  // h3 dead after conv1

  k_layernorm<<<Bc * CIN, 256, 0, stream>>>(x, ln_w, ln_b, h0);
  k_ca_kc<<<Bc * CIN, 256, 0, stream>>>(h0, ca_alpha, Kc);
  k_ca_softmax<<<Bc, 32, 0, stream>>>(Kc, ca_W, Ac);
  k_ca_apply<<<dim3(384, Bc), 256, 0, stream>>>(Ac, h0, h1);
  k_sa_ks<<<(Bc * NT + 255) / 256, 256, 0, stream>>>(h1, sa_alpha, Ks);
  k_sa_ksw<<<(Bc * Nn * Tt + 255) / 256, 256, 0, stream>>>(Ks, sa_W, KsW);
  k_sa_logits<<<dim3(512, Bc), 256, 0, stream>>>(KsW, Ks, As);
  k_sa_softmax<<<Bc * Nn, 256, 0, stream>>>(As, A);
  {
    const long long tot = (long long)SZ_CNT;
    k_transpose<<<(unsigned)((tot + 255) / 256), 256, 0, stream>>>(h1, hT);
  }
  // 1024/128 x 768/64 tile blocks, batched over b (async-LDS double buffered)
  k_gcn_agg<<<dim3(Nn / AGG_BM, (CIN * Tt) / AGG_BN, Bc), 256, 0, stream>>>(As, hT, g);
  k_gcn_mix<<<dim3(32, Bc, Tt), 256, 0, stream>>>(gcn_W, g, h2);
  k_ta_kt<<<(Bc * Tt * Nn + 255) / 256, 256, 0, stream>>>(h2, ta_alpha, Kt);
  k_ta_p<<<(Bc * Tt * 10 + 255) / 256, 256, 0, stream>>>(Kt, ta_W1, ta_W2, P1, P2);
  k_ta_softmax<<<Bc, 32, 0, stream>>>(P1, P2, At);
  k_ta_apply<<<dim3(1024, Bc), 256, 0, stream>>>(At, h2, h3);
  {
    const long long tot = (long long)SZ_ONT;
    const unsigned gb = (unsigned)((tot + 255) / 256);
    k_conv<<<gb, 256, 0, stream>>>(h3, c1w, c1b, h4, 1);
    k_conv<<<gb, 256, 0, stream>>>(h4, c2w, c2b, h5, 2);
    k_residual<<<gb, 256, 0, stream>>>(h5, x, rw, rb, out);
  }
}
